// HyperbolicAttention_13194139533842
// MI455X (gfx1250) — matrix operand order repaired and checked by reading, not compiled, not run on hardware
//
#include <hip/hip_runtime.h>
#include <hip/hip_bf16.h>
#include <stdint.h>

// ---------------------------------------------------------------------------
// Hyperbolic multi-head attention for MI455X (gfx1250), wave32 + WMMA.
// B=2, N=4096, D_IN=256, UNITS=512, H=8, depth=64, curvature c=1.
// ---------------------------------------------------------------------------

typedef __attribute__((ext_vector_type(16))) _Float16 v16h;
typedef __attribute__((ext_vector_type(8)))  float    v8f;

union AFrag {
    v16h     h;
    uint32_t u[8];
    _Float16 f[16];
};

#define BATCH   2
#define SEQ     4096
#define BN      (BATCH * SEQ)      // 8192
#define DIN     256
#define UNITS   512
#define HEADS   8
#define DEPTH   64
#define EPSF    1e-7f
#define MAXARG  (1.0f - 1e-5f)

// ---------------------------------------------------------------------------
// K0: weights -> f16, transposed (Bt[n][k] = B[k][n]) so WMMA B-fragments
// load contiguous K-pairs.  Wq|Wk|Wv concatenated into one [1536][256] Bt.
// ---------------------------------------------------------------------------
__global__ __launch_bounds__(256)
void prep_weights(const float* __restrict__ Wq, const float* __restrict__ Wk,
                  const float* __restrict__ Wv, const float* __restrict__ Wo,
                  _Float16* __restrict__ Wqkv_t, _Float16* __restrict__ Wo_t) {
    int idx = blockIdx.x * blockDim.x + threadIdx.x;
    if (idx < 1536 * 256) {
        int n = idx >> 8, k = idx & 255;          // n: output unit, k: input dim
        int w = n >> 9, u = n & 511;
        const float* W = (w == 0) ? Wq : ((w == 1) ? Wk : Wv);
        Wqkv_t[idx] = (_Float16)W[(size_t)k * 512 + u];
        return;
    }
    idx -= 1536 * 256;
    if (idx < 512 * 512) {
        int n = idx >> 9, k = idx & 511;
        Wo_t[idx] = (_Float16)Wo[(size_t)k * 512 + n];
    }
}

// ---------------------------------------------------------------------------
// K1: tangent = logmap0(inputs) over 256-wide rows, emitted as f16.
// ---------------------------------------------------------------------------
__global__ __launch_bounds__(256)
void logmap_rows(const float* __restrict__ x, _Float16* __restrict__ t16) {
    __shared__ float red[8];
    const int row = blockIdx.x, tid = threadIdx.x;
    float v = x[(size_t)row * DIN + tid];
    float s = v * v;
    #pragma unroll
    for (int m = 16; m >= 1; m >>= 1) s += __shfl_xor(s, m, 32);
    if ((tid & 31) == 0) red[tid >> 5] = s;
    __syncthreads();
    float tot = 0.f;
    #pragma unroll
    for (int i = 0; i < 8; ++i) tot += red[i];
    float n   = fmaxf(sqrtf(tot), EPSF);
    float arg = fminf(n, MAXARG);
    float scl = atanhf(arg) / n;
    t16[(size_t)row * DIN + tid] = (_Float16)(v * scl);
}

// ---------------------------------------------------------------------------
// Generic WMMA GEMM:  C[M,N] = A[M,K] * Bt[N,K]^T   (f16 in, f32 out).
// One 16x64 output tile per wave: per 32-wide K step, 1 A-fragment feeds
// 4 B-fragments / 4 v_wmma -> 2.5 b128 loads per WMMA.
// A 16x32 fragment per lane: row = lane&15, 16 contiguous K values (k = 16*hh + s).
// B 32x16 fragment per lane: col = lane&15, 16 contiguous K values.
// Requires N % 64 == 0, K % 32 == 0, M % 16 == 0.
// ---------------------------------------------------------------------------
__global__ __launch_bounds__(256)
void gemm_tn_f16(const _Float16* __restrict__ A, const _Float16* __restrict__ Bt,
                 float* __restrict__ C, int M, int N, int K) {
    const int lane = threadIdx.x & 31;
    const int wid  = blockIdx.x * (blockDim.x >> 5) + (threadIdx.x >> 5);
    const int nt64 = N >> 6;
    const int mt = wid / nt64, nb = wid % nt64;
    const int l15 = lane & 15, hh = lane >> 4;

    const uint32_t* Arow = (const uint32_t*)(A + (size_t)(mt * 16 + l15) * K);
    const uint32_t* Brow[4];
    #pragma unroll
    for (int j = 0; j < 4; ++j)
        Brow[j] = (const uint32_t*)(Bt + (size_t)(nb * 64 + j * 16 + l15) * K);

    v8f acc[4];
    #pragma unroll
    for (int j = 0; j < 4; ++j) acc[j] = (v8f){};

    for (int k0 = 0; k0 < K; k0 += 32) {
        AFrag a;
        #pragma unroll
        for (int v = 0; v < 8; ++v) {
            int ka = k0 + hh * 16 + 2 * v;
            a.u[v] = Arow[ka >> 1];
        }
        const int bb = (k0 >> 1) + hh * 8;        // contiguous 8 dwords per frag
        #pragma unroll
        for (int j = 0; j < 4; ++j) {
            AFrag b;
            #pragma unroll
            for (int v = 0; v < 8; ++v) b.u[v] = Brow[j][bb + v];
            acc[j] = __builtin_amdgcn_wmma_f32_16x16x32_f16(false, a.h, false, b.h,
                                                            (short)0, acc[j], false, false);
        }
    }
    // C layout: VGPR i -> row mt*16 + i + 8*hh, col = tile col + l15
    #pragma unroll
    for (int j = 0; j < 4; ++j) {
        float* Cp = C + (size_t)(mt * 16 + hh * 8) * N + nb * 64 + j * 16 + l15;
        #pragma unroll
        for (int i = 0; i < 8; ++i) Cp[(size_t)i * N] = acc[j][i];
    }
}

// ---------------------------------------------------------------------------
// K3: per (bn, head): q_hyp = expmap0(q), k_hyp = expmap0(k),
//     v_tan = logmap0(v), plus ||q_hyp||^2, ||k_hyp||^2 per row.
// qkv is the fused GEMM output: [BN][1536] = [q | k | v].
// ---------------------------------------------------------------------------
__global__ __launch_bounds__(64)
void qkv_maps(const float* __restrict__ qkv,
              _Float16* __restrict__ qh, _Float16* __restrict__ kh,
              _Float16* __restrict__ vt,
              float* __restrict__ q2, float* __restrict__ k2) {
    const int bnh = blockIdx.x;          // bn*8 + h
    const int bn = bnh >> 3, h = bnh & 7;
    const int d = threadIdx.x;           // 0..63
    const size_t base = (size_t)bn * 1536 + h * DEPTH + d;
    float qv = qkv[base], kv = qkv[base + 512], vv = qkv[base + 1024];

    __shared__ float red[3][64];
    red[0][d] = qv * qv; red[1][d] = kv * kv; red[2][d] = vv * vv;
    __syncthreads();
    for (int s = 32; s >= 1; s >>= 1) {
        if (d < s) {
            red[0][d] += red[0][d + s];
            red[1][d] += red[1][d + s];
            red[2][d] += red[2][d + s];
        }
        __syncthreads();
    }
    float sq = red[0][0], sk = red[1][0], sv = red[2][0];
    float nq = fmaxf(sqrtf(sq), EPSF);
    float nk = fmaxf(sqrtf(sk), EPSF);
    float nv = fmaxf(sqrtf(sv), EPSF);
    float fq = tanhf(nq) / nq;
    float fk = tanhf(nk) / nk;
    float fv = atanhf(fminf(nv, MAXARG)) / nv;

    const size_t ob = (size_t)bn * UNITS + h * DEPTH + d;
    qh[ob] = (_Float16)(qv * fq);
    kh[ob] = (_Float16)(kv * fk);
    vt[ob] = (_Float16)(vv * fv);
    if (d == 0) {
        q2[bnh] = fq * fq * sq;   // ||q_hyp||^2
        k2[bnh] = fk * fk * sk;
    }
}

// ---------------------------------------------------------------------------
// K3b: transpose v_tangent per (b,h) to depth-major vtT[(b*H+h)*64 + d][n]
// so flash-attention B-fragments (K = key index) are contiguous b128 loads.
// 64x64 tiles through padded LDS; coalesced on both sides.
// ---------------------------------------------------------------------------
__global__ __launch_bounds__(256)
void transpose_v(const _Float16* __restrict__ vt, _Float16* __restrict__ vtT) {
    __shared__ __align__(16) _Float16 tile[64][66];   // +2 pad: conflict-free
    const int blk = blockIdx.x;                        // b*H*(SEQ/64) blocks
    const int nt = blk & 63;           // SEQ/64 = 64 tiles along n
    const int h  = (blk >> 6) & 7;
    const int b  = blk >> 9;
    const int n0 = nt * 64;
    const int cr = threadIdx.x >> 6;   // 0..3
    const int cc = threadIdx.x & 63;
    #pragma unroll
    for (int it = 0; it < 16; ++it) {
        int r = it * 4 + cr;
        tile[r][cc] = vt[(size_t)(b * SEQ + n0 + r) * UNITS + h * DEPTH + cc];
    }
    __syncthreads();
    #pragma unroll
    for (int it = 0; it < 16; ++it) {
        int d = it * 4 + cr;
        vtT[((size_t)(b * HEADS + h) * DEPTH + d) * SEQ + n0 + cc] = tile[cc][d];
    }
}

// ---------------------------------------------------------------------------
// K4: flash attention.  One wave owns 16 queries of one (b,h) and streams all
// 4096 keys in 32-key blocks:
//   S  = Q * K^T      : 2 subtiles x 2 K-steps = 4 v_wmma (depth=64)
//   online softmax    : 16-lane shfl row reductions, scores stay in VGPRs
//   P relayout C->A   : per-wave LDS bounce (DS ops are in-order per wave)
//   ctx += P * Vtan   : 4 v_wmma, V read depth-major -> 2 b128 per fragment
// distances = (2*cross - q2 - k2) / sqrt(depth) folded into the score.
// Consecutive waves in a block take consecutive query tiles of the same
// (b,h), so the 8 waves share the K/V stream through WGP$.
// ---------------------------------------------------------------------------
__global__ __launch_bounds__(256)
void flash_attn(const _Float16* __restrict__ qh, const _Float16* __restrict__ kh,
                const _Float16* __restrict__ vtT, const float* __restrict__ q2,
                const float* __restrict__ k2, _Float16* __restrict__ ctx16) {
    __shared__ __align__(16) _Float16 lds[8][16 * 32];   // per-wave P tile
    const int lane = threadIdx.x & 31;
    const int w    = threadIdx.x >> 5;
    int wg = blockIdx.x * 8 + w;                 // 4096 waves total
    const int qt = wg & 255; wg >>= 8;
    const int h  = wg & 7;   const int b = wg >> 3;
    const int l15 = lane & 15, hh = lane >> 4;

    // Preload the 16x64 Q fragment pair (A-layout), kept live all iterations.
    AFrag qf[2];
    const uint32_t* Qp =
        (const uint32_t*)(qh + (size_t)(b * SEQ + qt * 16 + l15) * UNITS + h * DEPTH);
    #pragma unroll
    for (int dstep = 0; dstep < 2; ++dstep)
        #pragma unroll
        for (int v = 0; v < 8; ++v) {
            int kk = dstep * 32 + hh * 16 + 2 * v;
            qf[dstep].u[v] = Qp[kk >> 1];
        }

    float q2r[8];
    #pragma unroll
    for (int i = 0; i < 8; ++i)
        q2r[i] = q2[(size_t)(b * SEQ + qt * 16 + i + 8 * hh) * HEADS + h];

    // Depth-major V row bases for the 4 depth tiles (this lane's column).
    const uint32_t* Vrow[4];
    #pragma unroll
    for (int t = 0; t < 4; ++t)
        Vrow[t] = (const uint32_t*)(vtT +
                    ((size_t)(b * HEADS + h) * DEPTH + t * 16 + l15) * SEQ);

    float mi[8], li[8];
    v8f acc[4];
    #pragma unroll
    for (int i = 0; i < 8; ++i) { mi[i] = -1e30f; li[i] = 0.f; }
    #pragma unroll
    for (int t = 0; t < 4; ++t) acc[t] = (v8f){};
    _Float16* myLds = lds[w];

    for (int kb0 = 0; kb0 < SEQ; kb0 += 32) {
        // ---- scores for two 16-key subtiles --------------------------------
        float sc[2][8];
        #pragma unroll
        for (int st = 0; st < 2; ++st) {
            const int keyc = kb0 + st * 16 + l15;           // this lane's C column
            const uint32_t* Kp =
                (const uint32_t*)(kh + (size_t)(b * SEQ + keyc) * UNITS + h * DEPTH);
            v8f s = {};
            #pragma unroll
            for (int dstep = 0; dstep < 2; ++dstep) {
                AFrag bf;
                #pragma unroll
                for (int v = 0; v < 8; ++v) {
                    int kk = dstep * 32 + hh * 16 + 2 * v;  // B-frag: contiguous K
                    bf.u[v] = Kp[kk >> 1];
                }
                s = __builtin_amdgcn_wmma_f32_16x16x32_f16(false, qf[dstep].h, false,
                                                           bf.h, (short)0, s, false, false);
            }
            const float k2c = k2[(size_t)(b * SEQ + keyc) * HEADS + h];
            #pragma unroll
            for (int i = 0; i < 8; ++i)
                sc[st][i] = (2.f * s[i] - q2r[i] - k2c) * 0.125f;  // /sqrt(64)
        }
        // ---- online softmax (row = query = i + 8*hh, cols across 16 lanes) -
        float p0[8], p1[8];
        #pragma unroll
        for (int i = 0; i < 8; ++i) {
            float mx = fmaxf(sc[0][i], sc[1][i]);
            #pragma unroll
            for (int msk = 8; msk >= 1; msk >>= 1)
                mx = fmaxf(mx, __shfl_xor(mx, msk, 16));
            float mnew = fmaxf(mi[i], mx);
            float resc = __expf(mi[i] - mnew);
            float e0 = __expf(sc[0][i] - mnew);
            float e1 = __expf(sc[1][i] - mnew);
            float rs = e0 + e1;
            #pragma unroll
            for (int msk = 8; msk >= 1; msk >>= 1)
                rs += __shfl_xor(rs, msk, 16);
            li[i] = li[i] * resc + rs;
            mi[i] = mnew;
            p0[i] = e0; p1[i] = e1;
            #pragma unroll
            for (int t = 0; t < 4; ++t) acc[t][i] = acc[t][i] * resc;
        }
        // ---- P: C-layout -> LDS -> A-layout fragment -----------------------
        #pragma unroll
        for (int i = 0; i < 8; ++i) {
            myLds[(i + 8 * hh) * 32 + l15]      = (_Float16)p0[i];
            myLds[(i + 8 * hh) * 32 + 16 + l15] = (_Float16)p1[i];
        }
        asm volatile("" ::: "memory");   // keep DS writes before DS reads
        AFrag pf;
        const uint32_t* Lp = (const uint32_t*)myLds;
        #pragma unroll
        for (int v = 0; v < 8; ++v) {
            int kk = hh * 16 + 2 * v;
            pf.u[v] = Lp[(l15 * 32 + kk) >> 1];
        }
        // ---- ctx += P(16x32) * Vtan(32x64), depth-major V: 2 b128 / frag ---
        const int vb = (kb0 >> 1) + hh * 8;
        #pragma unroll
        for (int t = 0; t < 4; ++t) {
            AFrag bf;
            #pragma unroll
            for (int v = 0; v < 8; ++v) bf.u[v] = Vrow[t][vb + v];
            acc[t] = __builtin_amdgcn_wmma_f32_16x16x32_f16(false, pf.h, false, bf.h,
                                                            (short)0, acc[t], false, false);
        }
    }
    // ---- epilogue: normalize, merge heads into [BN][UNITS] f16 ------------
    #pragma unroll
    for (int i = 0; i < 8; ++i) {
        float inv = 1.f / li[i];
        size_t r = (size_t)(b * SEQ + qt * 16 + i + 8 * hh) * UNITS + h * DEPTH;
        #pragma unroll
        for (int t = 0; t < 4; ++t)
            ctx16[r + t * 16 + l15] = (_Float16)(acc[t][i] * inv);
    }
}

// ---------------------------------------------------------------------------
// K6: final expmap0 over 512-wide rows -> d_out (f32).
// ---------------------------------------------------------------------------
__global__ __launch_bounds__(256)
void expmap_out(const float* __restrict__ o, float* __restrict__ out) {
    __shared__ float red[8];
    const int row = blockIdx.x, tid = threadIdx.x;
    const float* r = o + (size_t)row * UNITS;
    float a = r[tid], c = r[tid + 256];
    float s = a * a + c * c;
    #pragma unroll
    for (int m = 16; m >= 1; m >>= 1) s += __shfl_xor(s, m, 32);
    if ((tid & 31) == 0) red[tid >> 5] = s;
    __syncthreads();
    float tot = 0.f;
    #pragma unroll
    for (int i = 0; i < 8; ++i) tot += red[i];
    float n   = fmaxf(sqrtf(tot), EPSF);
    float scl = tanhf(n) / n;
    out[(size_t)row * UNITS + tid]       = a * scl;
    out[(size_t)row * UNITS + tid + 256] = c * scl;
}

// ---------------------------------------------------------------------------
// Launcher.  Workspace (~98 MB; harness scratch assumed sufficient):
//   tan16[BN,256]f16 | Wqkv_t[1536,256]f16 | Wo_t[512,512]f16 |
//   qkv[BN,1536]f32 (reused for out[BN,512]f32 after consumed) |
//   qh/kh/vt[BN,512]f16 | vtT[B*H*64,4096]f16 | q2/k2[BN,8]f32 | ctx16 f16
// ---------------------------------------------------------------------------
extern "C" void kernel_launch(void* const* d_in, const int* in_sizes, int n_in,
                              void* d_out, int out_size, void* d_ws, size_t ws_size,
                              hipStream_t stream) {
    const float* inputs = (const float*)d_in[0];
    const float* Wq = (const float*)d_in[1];
    const float* Wk = (const float*)d_in[2];
    const float* Wv = (const float*)d_in[3];
    const float* Wo = (const float*)d_in[4];
    float* out = (float*)d_out;

    char* base = (char*)d_ws;
    size_t off = 0;
    auto take = [&](size_t bytes) -> char* {
        char* p = base + off;
        off = (off + bytes + 255) & ~(size_t)255;
        return p;
    };
    _Float16* tan16  = (_Float16*)take((size_t)BN * DIN * 2);
    _Float16* wqkv_t = (_Float16*)take((size_t)1536 * DIN * 2);
    _Float16* wo_t   = (_Float16*)take((size_t)UNITS * UNITS * 2);
    float*    qkv    = (float*)   take((size_t)BN * 1536 * 4);
    _Float16* qh16   = (_Float16*)take((size_t)BN * UNITS * 2);
    _Float16* kh16   = (_Float16*)take((size_t)BN * UNITS * 2);
    _Float16* vt16   = (_Float16*)take((size_t)BN * UNITS * 2);
    _Float16* vtT    = (_Float16*)take((size_t)BN * UNITS * 2);
    float*    q2     = (float*)   take((size_t)BN * HEADS * 4);
    float*    k2     = (float*)   take((size_t)BN * HEADS * 4);
    _Float16* ctx16  = (_Float16*)take((size_t)BN * UNITS * 2);
    float*    outf   = (float*)qkv;   // overlay: qkv f32 is dead after qkv_maps

    (void)in_sizes; (void)n_in; (void)out_size; (void)ws_size;

    // 0) weights -> f16 transposed
    {
        int total = 1536 * 256 + 512 * 512;
        prep_weights<<<(total + 255) / 256, 256, 0, stream>>>(Wq, Wk, Wv, Wo, wqkv_t, wo_t);
    }
    // 1) tangent = logmap0(inputs)
    logmap_rows<<<BN, 256, 0, stream>>>(inputs, tan16);
    // 2) fused QKV GEMM: [8192,256] x [256,1536], 16x64 tile per wave
    {
        int waves = (BN / 16) * (1536 / 64);      // 12288
        gemm_tn_f16<<<waves / 8, 256, 0, stream>>>(tan16, wqkv_t, qkv, BN, 1536, DIN);
    }
    // 3) exp/log maps + squared norms
    qkv_maps<<<BN * HEADS, 64, 0, stream>>>(qkv, qh16, kh16, vt16, q2, k2);
    // 3b) v_tangent -> depth-major layout for the attention PV GEMM
    transpose_v<<<BATCH * HEADS * (SEQ / 64), 256, 0, stream>>>(vt16, vtT);
    // 4) flash attention -> ctx16
    {
        int waves = BATCH * HEADS * (SEQ / 16);   // 4096 waves
        flash_attn<<<waves / 8, 256, 0, stream>>>(qh16, kh16, vtT, q2, k2, ctx16);
    }
    // 5) output GEMM: [8192,512] x [512,512], 16x64 tile per wave
    {
        int waves = (BN / 16) * (UNITS / 64);     // 4096
        gemm_tn_f16<<<waves / 8, 256, 0, stream>>>(ctx16, wo_t, outf, BN, UNITS, UNITS);
    }
    // 6) final expmap0 -> d_out
    expmap_out<<<BN, 256, 0, stream>>>(outf, out);
}
